// SliceAttention_36215164240649
// MI455X (gfx1250) — matrix operand order repaired and checked by reading, not compiled, not run on hardware
//
#include <hip/hip_runtime.h>
#include <cstdint>

// ---------------------------------------------------------------------------
// SliceAttention for MI455X (gfx1250, wave32, WMMA 16x16x32 bf16)
//
// Math:  sw = softmax(x@Wa + ba)             (B,N,64)
//        st = (sw^T @ x) / colsum           (B,64,768)
//        q,k,v = phi(st@Wq), phi(st@Wk), st@Wv ; linear attention per head
//        M  = out_slice @ Wo                (B,64,768)   [algebraic refactor]
//        out = sw @ M + bo                  (B,N,768)
// ---------------------------------------------------------------------------

#define BATCH   4
#define NTOK    32768
#define CH      768
#define NSLICE  64
#define NHEAD   12
#define HDIM    64
#define KCHUNK  16          // split-K chunks for K2 (2048 tokens each)

typedef __attribute__((ext_vector_type(16))) __bf16 v16bf;
typedef __attribute__((ext_vector_type(8)))  float  v8f;

__device__ __forceinline__ unsigned short f2bf(float f) {
    uint32_t u = __builtin_bit_cast(uint32_t, f);
    u += 0x7FFFu + ((u >> 16) & 1u);      // round-to-nearest-even
    return (unsigned short)(u >> 16);
}

// A fragment (16x32 bf16, M x K), LDS row-major [m][k], row stride ldm (even).
// lanes0-15 M=0..15; VGPR j -> K=half*16+2j (matches loadB k order).
__device__ __forceinline__ v16bf loadA(const unsigned short* lds, int mBase,
                                       int kBase, int ldm, int lane) {
    int m = lane & 15, half = lane >> 4;
    union { uint32_t u[8]; v16bf v; } f;
    const unsigned short* row = lds + (mBase + m) * ldm + kBase;
#pragma unroll
    for (int j = 0; j < 8; ++j) {
        int k = half * 16 + 2 * j;
        f.u[j] = *(const uint32_t*)(row + k);
    }
    return f.v;
}

// B fragment (32x16 bf16, K x N), LDS holds B^T row-major [n][k], row stride ldb.
// lanes0-15 K=0..15, lanes16-31 K=16..31; VGPR j -> K=khalf*16+2j (pair contiguous).
__device__ __forceinline__ v16bf loadB(const unsigned short* ldsT, int nBase,
                                       int kBase, int ldb, int lane) {
    int n = lane & 15, khalf = lane >> 4;
    union { uint32_t u[8]; v16bf v; } f;
    const unsigned short* row = ldsT + (nBase + n) * ldb + kBase + khalf * 16;
#pragma unroll
    for (int j = 0; j < 8; ++j) f.u[j] = *(const uint32_t*)(row + 2 * j);
    return f.v;
}

#define WMMA_BF16(a, b, c) \
    __builtin_amdgcn_wmma_f32_16x16x32_bf16(false, (a), false, (b), (short)0, (c), false, false)

// ---------------------------------------------------------------------------
__global__ void k0_zero(float* p, int n) {
    int i = blockIdx.x * 256 + threadIdx.x;
    if (i < n) p[i] = 0.0f;
}

// ---------------------------------------------------------------------------
// K1: logits = x@Wa + ba over 128-token tiles (WMMA), softmax -> sw, colsum.
__global__ __launch_bounds__(256) void k1_assign(
    const float* __restrict__ x, const float* __restrict__ Wa,
    const float* __restrict__ ba, float* __restrict__ sw,
    float* __restrict__ colsum) {
    __shared__ unsigned short xS[128 * 32];   // [token][k] bf16
    __shared__ unsigned short wT[64 * 32];    // [slice][k] bf16 (B^T)
    __shared__ float logits[128 * 64];

    int tid = threadIdx.x, lane = tid & 31, w = tid >> 5;
    long row0 = (long)blockIdx.x * 128;

    v8f acc[4];
#pragma unroll
    for (int t = 0; t < 4; ++t) acc[t] = (v8f){};

    for (int kk = 0; kk < CH; kk += 32) {
        // prefetch next k-tile of x (hides HBM latency across the barrier)
        if (kk + 32 < CH)
            __builtin_prefetch(&x[(row0 + (tid >> 1)) * CH + kk + 32 + (tid & 1) * 16], 0, 1);
#pragma unroll
        for (int i = 0; i < 16; ++i) {
            int idx = tid + i * 256;               // 4096 = 128x32
            int r = idx >> 5, c = idx & 31;
            xS[idx] = f2bf(x[(row0 + r) * CH + kk + c]);
        }
#pragma unroll
        for (int i = 0; i < 8; ++i) {
            int idx = tid + i * 256;               // 2048 = 32x64
            int k = idx >> 6, c = idx & 63;
            wT[c * 32 + k] = f2bf(Wa[(kk + k) * NSLICE + c]);
        }
        __syncthreads();
        v16bf af = loadA(xS, w * 16, 0, 32, lane);
#pragma unroll
        for (int t = 0; t < 4; ++t) {
            v16bf bf = loadB(wT, t * 16, 0, 32, lane);
            acc[t] = WMMA_BF16(af, bf, acc[t]);
        }
        __syncthreads();
    }
    int nl = lane & 15, mh = (lane >> 4) * 8;
#pragma unroll
    for (int t = 0; t < 4; ++t)
#pragma unroll
        for (int r = 0; r < 8; ++r)
            logits[(w * 16 + r + mh) * 64 + t * 16 + nl] = acc[t][r];
    __syncthreads();

    if (tid < 128) {
        float* L = &logits[tid * 64];
        float mx = -3.4e38f;
        for (int s = 0; s < 64; ++s) { float t = L[s] + ba[s]; L[s] = t; mx = t > mx ? t : mx; }
        float sum = 0.0f;
        for (int s = 0; s < 64; ++s) { float e = expf(L[s] - mx); L[s] = e; sum += e; }
        float inv = 1.0f / sum;
        for (int s = 0; s < 64; ++s) {
            float p = L[s] * inv;
            L[s] = p;
            sw[(row0 + tid) * NSLICE + s] = p;
        }
    }
    __syncthreads();
    if (tid < 64) {
        float s = 0.0f;
        for (int r = 0; r < 128; ++r) s += logits[r * 64 + tid];
        int b = (int)(row0 >> 15);                 // N = 32768
        atomicAdd(&colsum[b * NSLICE + tid], s);
    }
}

// ---------------------------------------------------------------------------
// K2: part[chunk][b][s][c] = sum_{n in chunk} sw[b][n][s] * x[b][n][c]
//     (split-K over tokens, no atomics; reduced by k2r)
__global__ __launch_bounds__(256) void k2_slicetok(
    const float* __restrict__ sw, const float* __restrict__ x,
    float* __restrict__ part) {
    __shared__ unsigned short swT[64 * 32];   // [slice][token] (A row-major)
    __shared__ unsigned short xT[64 * 32];    // [chan][token]  (B^T)

    int tid = threadIdx.x, lane = tid & 31, w = tid >> 5;
    int c0    = blockIdx.x * 64;
    int chunk = blockIdx.y;
    int b     = blockIdx.z;
    long rowBase = (long)b * NTOK + chunk * (NTOK / KCHUNK);

    int sRow  = (w & 3) * 16;
    int cPair = (w >> 2);
    v8f av[2]; av[0] = (v8f){}; av[1] = (v8f){};

    for (int t0 = 0; t0 < NTOK / KCHUNK; t0 += 32) {
        if (t0 + 32 < NTOK / KCHUNK)
            __builtin_prefetch(&x[(rowBase + t0 + 32 + (tid >> 3)) * CH + c0 + (tid & 7) * 8], 0, 1);
#pragma unroll
        for (int i = 0; i < 8; ++i) {
            int idx = tid + i * 256;              // 2048 = 32x64
            int t = idx >> 6, s = idx & 63;
            swT[s * 32 + t] = f2bf(sw[(rowBase + t0 + t) * NSLICE + s]);
        }
#pragma unroll
        for (int i = 0; i < 8; ++i) {
            int idx = tid + i * 256;
            int t = idx >> 6, c = idx & 63;
            xT[c * 32 + t] = f2bf(x[(rowBase + t0 + t) * CH + c0 + c]);
        }
        __syncthreads();
        v16bf af = loadA(swT, sRow, 0, 32, lane);
#pragma unroll
        for (int j = 0; j < 2; ++j) {
            v16bf bf = loadB(xT, (cPair * 2 + j) * 16, 0, 32, lane);
            av[j] = WMMA_BF16(af, bf, av[j]);
        }
        __syncthreads();
    }
    int nl = lane & 15, mh = (lane >> 4) * 8;
    long base = ((long)chunk * BATCH + b) * NSLICE * CH;
#pragma unroll
    for (int j = 0; j < 2; ++j)
#pragma unroll
        for (int r = 0; r < 8; ++r) {
            int s = sRow + r + mh;
            int c = c0 + (cPair * 2 + j) * 16 + nl;
            part[base + (long)s * CH + c] = av[j][r];
        }
}

// k2r: st[b][s][c] = (sum_chunk part) / max(colsum, 1e-6)   (fixed sum order)
__global__ void k2r_reduce(const float* __restrict__ part,
                           const float* __restrict__ colsum,
                           float* __restrict__ st) {
    const long SC = (long)BATCH * NSLICE * CH;
    for (long i = (long)blockIdx.x * 256 + threadIdx.x; i < SC;
         i += (long)gridDim.x * 256) {
        float s = 0.0f;
#pragma unroll
        for (int c = 0; c < KCHUNK; ++c) s += part[(long)c * SC + i];
        int b  = (int)(i / (NSLICE * CH));
        int sl = (int)((i / CH) % NSLICE);
        float d = colsum[b * NSLICE + sl];
        d = d < 1e-6f ? 1e-6f : d;
        st[i] = s / d;
    }
}

// ---------------------------------------------------------------------------
// K3b: q/k/v = st @ W ; ELU+1 feature map on q,k.
__global__ __launch_bounds__(256) void k3b_qkv(
    const float* __restrict__ st, const float* __restrict__ Wq,
    const float* __restrict__ Wk, const float* __restrict__ Wv,
    float* __restrict__ q, float* __restrict__ k, float* __restrict__ v) {
    __shared__ float aS[64 * 64];
    __shared__ float wS[64 * 64];
    int b = blockIdx.x, p = blockIdx.y, h = blockIdx.z;
    const float* W = (p == 0) ? Wq : (p == 1) ? Wk : Wv;
    float* outp    = (p == 0) ? q  : (p == 1) ? k  : v;
    int c0 = h * HDIM, tid = threadIdx.x;
    int s = tid >> 2, cg = (tid & 3) * 16;
    float accr[16];
#pragma unroll
    for (int i = 0; i < 16; ++i) accr[i] = 0.0f;

    for (int kk = 0; kk < CH; kk += 64) {
        for (int i = tid; i < 4096; i += 256) {
            int r = i >> 6, cc = i & 63;
            aS[i] = st[(long)b * NSLICE * CH + r * CH + kk + cc];
            wS[i] = W[(kk + r) * CH + c0 + cc];
        }
        __syncthreads();
        for (int j = 0; j < 64; ++j) {
            float a = aS[s * 64 + j];
#pragma unroll
            for (int i = 0; i < 16; ++i) accr[i] += a * wS[j * 64 + cg + i];
        }
        __syncthreads();
    }
#pragma unroll
    for (int i = 0; i < 16; ++i) {
        float val = accr[i];
        if (p < 2) val = (val > 0.0f) ? (val + 1.0f) : expf(val);   // elu(x)+1
        outp[(long)b * NSLICE * CH + s * CH + c0 + cg + i] = val;
    }
}

// K3c: linear attention per (batch, head), S = 64.
__global__ __launch_bounds__(64) void k3c_attn(
    const float* __restrict__ q, const float* __restrict__ k,
    const float* __restrict__ v, float* __restrict__ os) {
    __shared__ float qS[64 * 64], kS[64 * 64], kvS[64 * 64];
    __shared__ float ksum[64];
    int b = blockIdx.x, h = blockIdx.y, tid = threadIdx.x;
    long base = (long)b * NSLICE * CH + h * HDIM;

    for (int i = tid; i < 4096; i += 64) {
        int s = i >> 6, d = i & 63;
        qS[i] = q[base + s * CH + d];
        kS[i] = k[base + s * CH + d];
    }
    __syncthreads();
    float ks = 0.0f;
    for (int s = 0; s < 64; ++s) ks += kS[s * 64 + tid];
    ksum[tid] = ks;
    float vloc[64];
    for (int s = 0; s < 64; ++s) vloc[s] = v[base + s * CH + tid];
    for (int d = 0; d < 64; ++d) {
        float a = 0.0f;
        for (int s = 0; s < 64; ++s) a += kS[s * 64 + d] * vloc[s];
        kvS[d * 64 + tid] = a;
    }
    __syncthreads();
    float den = 0.0f;
    for (int d = 0; d < 64; ++d) den += qS[tid * 64 + d] * ksum[d];
    den = den < 1e-6f ? 1e-6f : den;
    float inv = 1.0f / den;
    for (int e = 0; e < 64; ++e) {
        float num = 0.0f;
        for (int d = 0; d < 64; ++d) num += qS[tid * 64 + d] * kvS[d * 64 + e];
        os[base + tid * CH + e] = num * inv;
    }
}

// K3d: M = out_slice @ Wo   (b_o deferred to K4)
__global__ __launch_bounds__(256) void k3d_mproj(
    const float* __restrict__ os, const float* __restrict__ Wo,
    float* __restrict__ M) {
    __shared__ float aS[64 * 64];
    __shared__ float wS[64 * 64];
    int b = blockIdx.x, h = blockIdx.y;
    int c0 = h * HDIM, tid = threadIdx.x;
    int s = tid >> 2, cg = (tid & 3) * 16;
    float accr[16];
#pragma unroll
    for (int i = 0; i < 16; ++i) accr[i] = 0.0f;
    for (int kk = 0; kk < CH; kk += 64) {
        for (int i = tid; i < 4096; i += 256) {
            int r = i >> 6, cc = i & 63;
            aS[i] = os[(long)b * NSLICE * CH + r * CH + kk + cc];
            wS[i] = Wo[(kk + r) * CH + c0 + cc];
        }
        __syncthreads();
        for (int j = 0; j < 64; ++j) {
            float a = aS[s * 64 + j];
#pragma unroll
            for (int i = 0; i < 16; ++i) accr[i] += a * wS[j * 64 + cg + i];
        }
        __syncthreads();
    }
#pragma unroll
    for (int i = 0; i < 16; ++i)
        M[(long)b * NSLICE * CH + s * CH + c0 + cg + i] = accr[i];
}

// ---------------------------------------------------------------------------
// K4: out = sw @ M + bo   (M=tokens x 768, K=64) — WMMA, 128x128 tiles.
__global__ __launch_bounds__(256) void k4_out(
    const float* __restrict__ sw, const float* __restrict__ M,
    const float* __restrict__ bo, float* __restrict__ out) {
    __shared__ unsigned short swS[128 * 64];  // [token][slice] (A row-major)
    __shared__ unsigned short Mt[128 * 64];   // [chan][slice]  (B^T)
    __shared__ float boS[128];

    int tid = threadIdx.x, lane = tid & 31, w = tid >> 5;
    long n0 = (long)blockIdx.x * 128;
    int c0 = blockIdx.y * 128;
    int b  = blockIdx.z;
    long rowBase = (long)b * NTOK + n0;

#pragma unroll
    for (int i = 0; i < 32; ++i) {
        int idx = tid + i * 256;                  // 8192 = 128x64
        swS[idx] = f2bf(sw[(rowBase + (idx >> 6)) * NSLICE + (idx & 63)]);
    }
#pragma unroll
    for (int i = 0; i < 32; ++i) {
        int idx = tid + i * 256;
        int kk = idx >> 7, c = idx & 127;
        Mt[c * 64 + kk] = f2bf(M[((long)b * NSLICE + kk) * CH + c0 + c]);
    }
    if (tid < 128) boS[tid] = bo[c0 + tid];
    __syncthreads();

    v8f av[8];
#pragma unroll
    for (int t = 0; t < 8; ++t) av[t] = (v8f){};
#pragma unroll
    for (int kk = 0; kk < 64; kk += 32) {
        v16bf af = loadA(swS, w * 16, kk, 64, lane);
#pragma unroll
        for (int t = 0; t < 8; ++t) {
            v16bf bf = loadB(Mt, t * 16, kk, 64, lane);
            av[t] = WMMA_BF16(af, bf, av[t]);
        }
    }
    int nl = lane & 15, mh = (lane >> 4) * 8;
#pragma unroll
    for (int t = 0; t < 8; ++t)
#pragma unroll
        for (int r = 0; r < 8; ++r) {
            int m = w * 16 + r + mh;
            int c = t * 16 + nl;
            out[(rowBase + m) * CH + c0 + c] = av[t][r] + boS[c];
        }
}

// ---------------------------------------------------------------------------
extern "C" void kernel_launch(void* const* d_in, const int* in_sizes, int n_in,
                              void* d_out, int out_size, void* d_ws, size_t ws_size,
                              hipStream_t stream) {
    const float* x  = (const float*)d_in[0];
    const float* Wa = (const float*)d_in[1];
    const float* ba = (const float*)d_in[2];
    const float* Wq = (const float*)d_in[3];
    const float* Wk = (const float*)d_in[4];
    const float* Wv = (const float*)d_in[5];
    const float* Wo = (const float*)d_in[6];
    const float* bo = (const float*)d_in[7];
    float* out = (float*)d_out;
    float* ws  = (float*)d_ws;

    const long SWN = (long)BATCH * NTOK * NSLICE;      // 8388608
    const long SC  = (long)BATCH * NSLICE * CH;        // 196608
    float* sw     = ws;
    float* part   = sw + SWN;                          // KCHUNK * SC
    float* colsum = part + (long)KCHUNK * SC;
    float* st     = colsum + BATCH * NSLICE;
    float* qb     = st + SC;
    float* kb     = qb + SC;
    float* vb     = kb + SC;
    float* osb    = vb + SC;
    float* Mb     = osb + SC;

    k0_zero<<<dim3(1), dim3(256), 0, stream>>>(colsum, BATCH * NSLICE);
    k1_assign<<<dim3(BATCH * NTOK / 128), dim3(256), 0, stream>>>(x, Wa, ba, sw, colsum);
    k2_slicetok<<<dim3(CH / 64, KCHUNK, BATCH), dim3(256), 0, stream>>>(sw, x, part);
    k2r_reduce<<<dim3(768), dim3(256), 0, stream>>>(part, colsum, st);
    k3b_qkv<<<dim3(BATCH, 3, NHEAD), dim3(256), 0, stream>>>(st, Wq, Wk, Wv, qb, kb, vb);
    k3c_attn<<<dim3(BATCH, NHEAD), dim3(64), 0, stream>>>(qb, kb, vb, osb);
    k3d_mproj<<<dim3(BATCH, NHEAD), dim3(256), 0, stream>>>(osb, Wo, Mb);
    k4_out<<<dim3(NTOK / 128, CH / 128, BATCH), dim3(256), 0, stream>>>(sw, Mb, bo, out);
}
